// EdgeConvKNN_28303834480930
// MI455X (gfx1250) — matrix operand order repaired and checked by reading, not compiled, not run on hardware
//
#include <hip/hip_runtime.h>
#include <hip/hip_bf16.h>

typedef __attribute__((ext_vector_type(16))) _Float16 v16h;
typedef __attribute__((ext_vector_type(8)))  _Float16 v8h;
typedef __attribute__((ext_vector_type(8)))  float    v8f;
typedef __attribute__((ext_vector_type(4)))  float    v4f;   // native vec for NT stores

#define NB   32      // B = bsz * NHEAD
#define NN   1024    // N = L
#define CC   64      // hd
#define KNN  10
#define DM   512
#define BSZ  4
#define EPSB 1e-5f

// ---------------------------------------------------------------------------
// Wave-local LDS sync: DS ops from one wave complete in order (DScnt), so for
// cross-lane LDS traffic inside a single wave we only need to (a) stop the
// compiler from reordering LDS accesses across this point and (b) drain DScnt.
// No s_barrier needed — the LDS tiles below are wave-private.
// ---------------------------------------------------------------------------
__device__ __forceinline__ void wave_lds_sync() {
  __builtin_amdgcn_wave_barrier();
  asm volatile("s_wait_dscnt 0x0" ::: "memory");
  __builtin_amdgcn_wave_barrier();
}

// ---------------------------------------------------------------------------
// WMMA helpers (CDNA5 v_wmma_f32_16x16x32_f16, wave32)
// ---------------------------------------------------------------------------
__device__ __forceinline__ v8f wmma_f16(v16h a, v16h b, v8f c) {
  return __builtin_amdgcn_wmma_f32_16x16x32_f16(
      /*neg_a=*/false, a, /*neg_b=*/false, b,
      /*c_mod=*/(short)0, c, /*reuse_a=*/false, /*reuse_b=*/false);
}

// A-matrix 16x32 f16 fragment: lane holds 16 contiguous K elements of its
// row: K in [16h, 16h+16), h = lane>>4 (matches B fragment order).
__device__ __forceinline__ v16h load_fragA(const _Float16* __restrict__ Xb,
                                           int rowbase, int kc, int lr, int half) {
  const _Float16* p = Xb + (size_t)(rowbase + lr) * CC + kc + 16 * half;
  return *(const v16h*)p;
}

// B-matrix 32x16 f16 fragment: lane holds 16 contiguous K elements of
// column (lane&15): K in [16h, 16h+16). Column n of X·X^T's B == row n of X.
__device__ __forceinline__ v16h load_fragB_rows(const _Float16* __restrict__ Xb,
                                                int colbase, int kc, int lr, int half) {
  const _Float16* p = Xb + (size_t)(colbase + lr) * CC + kc + 16 * half;
  return *(const v16h*)p;
}

// ---------------------------------------------------------------------------
// Kernel 1: gather xt (B,N,C) from x (L,bsz,d_model), convert to f16, and
// compute sq[b,n] = sum_c f16(x)^2 (f32 accumulate) for consistent distances.
// ---------------------------------------------------------------------------
__launch_bounds__(256)
__global__ void pack_kernel(const float* __restrict__ x,
                            _Float16* __restrict__ Xh,
                            float* __restrict__ sq) {
  int tid = blockIdx.x * blockDim.x + threadIdx.x;   // one thread per (b,n)
  int b = tid >> 10, n = tid & 1023;
  // xt[b][n][c] = x[(n&255)*4 + (b>>3)][n>>8][(b&7)*64 + c]
  int l   = ((n & 255) << 2) | (b >> 3);
  int bz  = n >> 8;
  int cf0 = (b & 7) << 6;
  const float* src = x + ((size_t)l * (BSZ * DM) + (size_t)bz * DM + cf0);
  _Float16* dst = Xh + ((size_t)tid << 6);
  float s = 0.f;
  #pragma unroll
  for (int c = 0; c < CC; c += 8) {
    float4 a = *(const float4*)(src + c);
    float4 b4 = *(const float4*)(src + c + 4);
    v8h h;
    h[0] = (_Float16)a.x;  h[1] = (_Float16)a.y;
    h[2] = (_Float16)a.z;  h[3] = (_Float16)a.w;
    h[4] = (_Float16)b4.x; h[5] = (_Float16)b4.y;
    h[6] = (_Float16)b4.z; h[7] = (_Float16)b4.w;
    #pragma unroll
    for (int t = 0; t < 8; ++t) { float f = (float)h[t]; s += f * f; }
    *(v8h*)(dst + c) = h;
  }
  sq[tid] = s;
}

// ---------------------------------------------------------------------------
// Kernel 2: fused Gram-matrix WMMA + top-k. One wave owns 16 rows of one
// batch; streams 64 column tiles; pd = 2*inner - sq_i - sq_j; per-row top-10
// kept in registers (lane = row half: lanes 0-15 cols [0,8), 16-31 cols [8,16)
// of each tile), merged through wave-private LDS at the end. Barrier-free:
// waves are fully independent, LDS sync is wave-local (s_wait_dscnt).
// ---------------------------------------------------------------------------
__launch_bounds__(128)
__global__ void knn_kernel(const _Float16* __restrict__ Xh,
                           const float* __restrict__ sq,
                           int* __restrict__ knn) {
  __shared__ float tile[4][16][17];
  __shared__ float mv[4][32][KNN];
  __shared__ int   mi[4][32][KNN];

  int wv   = threadIdx.x >> 5;
  int lane = threadIdx.x & 31;
  int wid  = blockIdx.x * 4 + wv;        // 0..2047
  int bb   = wid >> 6;                   // batch
  int m0   = (wid & 63) << 4;            // row-strip base
  int half = lane >> 4, lr = lane & 15;

  const _Float16* Xb = Xh + (size_t)bb * NN * CC;

  v16h a0 = load_fragA(Xb, m0, 0,  lr, half);
  v16h a1 = load_fragA(Xb, m0, 32, lr, half);

  float sqr[8];
  #pragma unroll
  for (int r = 0; r < 8; ++r) sqr[r] = sq[bb * NN + m0 + r + 8 * half];

  float kv[KNN]; int ki[KNN];            // sorted descending, kv[9] = min
  #pragma unroll
  for (int i = 0; i < KNN; ++i) { kv[i] = -3.0e38f; ki[i] = 0; }

  for (int n0 = 0; n0 < NN; n0 += 16) {
    v16h b0 = load_fragB_rows(Xb, n0, 0,  lr, half);
    v16h b1 = load_fragB_rows(Xb, n0, 32, lr, half);
    v8f acc = {};
    acc = wmma_f16(a0, b0, acc);
    acc = wmma_f16(a1, b1, acc);

    float sqB = sq[bb * NN + n0 + lr];
    #pragma unroll
    for (int r = 0; r < 8; ++r) {
      float pd = 2.f * acc[r] - sqr[r] - sqB;   // element (M=r+8h, N=lr)
      tile[wv][r + 8 * half][lr] = pd;
    }
    wave_lds_sync();
    // lane scans row lr, cols [8*half, 8*half+8) of this tile
    #pragma unroll
    for (int t = 0; t < 8; ++t) {
      float v = tile[wv][lr][8 * half + t];
      int  idx = n0 + 8 * half + t;
      if (v > kv[KNN - 1]) {
        float cv = v; int ci = idx;
        #pragma unroll
        for (int p = 0; p < KNN; ++p) {
          if (cv > kv[p]) {
            float tf = kv[p]; kv[p] = cv; cv = tf;
            int   ti = ki[p]; ki[p] = ci; ci = ti;
          }
        }
      }
    }
    wave_lds_sync();   // loads drained before next tile overwrites
  }

  // merge the two half-lane lists of each row (wave-local exchange)
  #pragma unroll
  for (int i = 0; i < KNN; ++i) { mv[wv][lane][i] = kv[i]; mi[wv][lane][i] = ki[i]; }
  wave_lds_sync();
  if (half == 0) {
    #pragma unroll
    for (int i = 0; i < KNN; ++i) {
      float v = mv[wv][lane + 16][i]; int idx = mi[wv][lane + 16][i];
      if (v > kv[KNN - 1]) {
        float cv = v; int ci = idx;
        #pragma unroll
        for (int p = 0; p < KNN; ++p) {
          if (cv > kv[p]) {
            float tf = kv[p]; kv[p] = cv; cv = tf;
            int   ti = ki[p]; ki[p] = ci; ci = ti;
          }
        }
      }
    }
    int* dst = knn + (size_t)(bb * NN + m0 + lr) * KNN;
    #pragma unroll
    for (int i = 0; i < KNN; ++i) dst[i] = ki[i];
  }
}

// ---------------------------------------------------------------------------
// Kernel 3: P = X·W1a^T, Q = X·(W1b-W1a)^T  (per batch, WMMA f16->f32).
// w1 is (64, 128) row-major f32: column o of the GEMM's B matrix is 16
// contiguous entries of w1[o][*], converted to f16 in registers.
// P/Q are stored with default (RT) hints: they are re-read through L2.
// ---------------------------------------------------------------------------
__launch_bounds__(128)
__global__ void pq_kernel(const _Float16* __restrict__ Xh,
                          const float* __restrict__ w1,
                          float* __restrict__ P,
                          float* __restrict__ Q) {
  int wv   = threadIdx.x >> 5;
  int lane = threadIdx.x & 31;
  int wid  = blockIdx.x * 4 + wv;        // 0..16383
  int which = wid >> 13;                 // 0 = P, 1 = Q
  int bb = (wid >> 8) & 31;
  int t  = wid & 255;
  int n0 = (t >> 2) << 4;
  int o0 = (t & 3) << 4;
  int half = lane >> 4, lr = lane & 15;

  const _Float16* Xb = Xh + (size_t)bb * NN * CC;
  v16h a0 = load_fragA(Xb, n0, 0,  lr, half);
  v16h a1 = load_fragA(Xb, n0, 32, lr, half);

  const float* wrow = w1 + (size_t)(o0 + lr) * (2 * CC);
  v16h b0, b1;
  #pragma unroll
  for (int tt = 0; tt < 16; ++tt) {
    int k0 = 16 * half + tt;       // kc = 0 chunk
    int k1 = 32 + 16 * half + tt;  // kc = 32 chunk
    float f0 = which ? (wrow[64 + k0] - wrow[k0]) : wrow[k0];
    float f1 = which ? (wrow[64 + k1] - wrow[k1]) : wrow[k1];
    b0[tt] = (_Float16)f0;
    b1[tt] = (_Float16)f1;
  }

  v8f acc = {};
  acc = wmma_f16(a0, b0, acc);
  acc = wmma_f16(a1, b1, acc);

  float* out = (which ? Q : P) + (size_t)bb * NN * CC;
  #pragma unroll
  for (int r = 0; r < 8; ++r)
    out[(size_t)(n0 + r + 8 * half) * CC + o0 + lr] = acc[r];
}

// ---------------------------------------------------------------------------
// Kernel 4: per (b,n): h[k,c] = leaky(bn1(P[j,c] + Q[n,c])),
// x_out[c] = max_k h, e[k] = leaky(bn2(sum_c h*w2[c])).
// x_out is final output -> non-temporal stores; evals is re-read -> RT.
// ---------------------------------------------------------------------------
__launch_bounds__(256)
__global__ void point_kernel(const float* __restrict__ P, const float* __restrict__ Q,
                             const int* __restrict__ knn,
                             const float* __restrict__ g1, const float* __restrict__ b1,
                             const float* __restrict__ m1, const float* __restrict__ v1,
                             const float* __restrict__ w2, const float* __restrict__ g2,
                             const float* __restrict__ b2f, const float* __restrict__ m2,
                             const float* __restrict__ v2,
                             float* __restrict__ xout, float* __restrict__ evals) {
  __shared__ float s1s[CC], t1s[CC], w2s[CC];
  if (threadIdx.x < CC) {
    float s = g1[threadIdx.x] * rsqrtf(v1[threadIdx.x] + EPSB);
    s1s[threadIdx.x] = s;
    t1s[threadIdx.x] = b1[threadIdx.x] - m1[threadIdx.x] * s;
    w2s[threadIdx.x] = w2[threadIdx.x];
  }
  __syncthreads();
  float s2 = g2[0] * rsqrtf(v2[0] + EPSB);
  float t2 = b2f[0] - m2[0] * s2;

  int tid = blockIdx.x * blockDim.x + threadIdx.x;  // (b,n)
  int bb = tid >> 10;
  const float* qrow = Q + ((size_t)tid << 6);
  float qreg[CC];
  #pragma unroll
  for (int c = 0; c < CC; c += 4) {
    float4 v = *(const float4*)(qrow + c);
    qreg[c] = v.x; qreg[c + 1] = v.y; qreg[c + 2] = v.z; qreg[c + 3] = v.w;
  }
  float mx[CC];
  #pragma unroll
  for (int c = 0; c < CC; ++c) mx[c] = -3.0e38f;

  const int* kn = knn + (size_t)tid * KNN;
  const float* Pb = P + ((size_t)bb << 16);   // bb * N * C

  for (int kk = 0; kk < KNN; ++kk) {
    int j = kn[kk];
    const float* prow = Pb + ((size_t)j << 6);
    float acc = 0.f;
    #pragma unroll
    for (int c = 0; c < CC; c += 4) {
      float4 pv = *(const float4*)(prow + c);
      float y, h;
      y = s1s[c]     * (pv.x + qreg[c])     + t1s[c];     h = y >= 0.f ? y : 0.2f * y; mx[c]     = fmaxf(mx[c],     h); acc += h * w2s[c];
      y = s1s[c + 1] * (pv.y + qreg[c + 1]) + t1s[c + 1]; h = y >= 0.f ? y : 0.2f * y; mx[c + 1] = fmaxf(mx[c + 1], h); acc += h * w2s[c + 1];
      y = s1s[c + 2] * (pv.z + qreg[c + 2]) + t1s[c + 2]; h = y >= 0.f ? y : 0.2f * y; mx[c + 2] = fmaxf(mx[c + 2], h); acc += h * w2s[c + 2];
      y = s1s[c + 3] * (pv.w + qreg[c + 3]) + t1s[c + 3]; h = y >= 0.f ? y : 0.2f * y; mx[c + 3] = fmaxf(mx[c + 3], h); acc += h * w2s[c + 3];
    }
    float ye = s2 * acc + t2;
    evals[(size_t)tid * KNN + kk] = ye >= 0.f ? ye : 0.2f * ye;
  }
  float* xo = xout + ((size_t)tid << 6);
  #pragma unroll
  for (int c = 0; c < CC; c += 4) {
    v4f v = { mx[c], mx[c + 1], mx[c + 2], mx[c + 3] };
    __builtin_nontemporal_store(v, (v4f*)(xo + c));
  }
}

// ---------------------------------------------------------------------------
// Kernel 5: softmax + mask, one wave per row. Row = 1014 ones + 10 e-values,
// so the reduction needs only the 10 e's; non-temporal float4 fill (write-once
// 268 MB stream, keep it out of L2), then 10-lane scalar fixup. Same-wave
// stores are ordered (stores in order with stores), so fixup lands last.
// ---------------------------------------------------------------------------
__launch_bounds__(256)
__global__ void softmax_kernel(const float* __restrict__ evals,
                               const int* __restrict__ knn,
                               float* __restrict__ eij, float* __restrict__ mask) {
  int wv   = threadIdx.x >> 5;
  int lane = threadIdx.x & 31;
  int row  = blockIdx.x * 8 + wv;        // 0..32767 == (b,n)

  const float* e  = evals + (size_t)row * KNN;
  const int*   kn = knn   + (size_t)row * KNN;
  float ev = (lane < KNN) ? e[lane] : -3.0e38f;
  int   ji = (lane < KNN) ? kn[lane] : 0;

  float m = fmaxf(ev, 1.0f);             // the 1014 fill entries are 1.0
  #pragma unroll
  for (int s = 16; s > 0; s >>= 1) m = fmaxf(m, __shfl_xor(m, s, 32));
  float ex = (lane < KNN) ? __expf(ev - m) : 0.f;
  float ssum = ex;
  #pragma unroll
  for (int s = 16; s > 0; s >>= 1) ssum += __shfl_xor(ssum, s, 32);

  float base_e = __expf(1.0f - m);
  float denom  = (float)(NN - KNN) * base_e + ssum;
  float fillv  = base_e / denom;

  float* erow = eij  + ((size_t)row << 10);
  float* mrow = mask + ((size_t)row << 10);
  v4f fv = { fillv, fillv, fillv, fillv };
  v4f ov = { 1.f, 1.f, 1.f, 1.f };
  #pragma unroll
  for (int i = 0; i < 8; ++i) {
    int q = i * 32 + lane;
    __builtin_nontemporal_store(fv, (v4f*)(erow + q * 4));
    __builtin_nontemporal_store(ov, (v4f*)(mrow + q * 4));
  }
  if (lane < KNN) {
    erow[ji] = ex / denom;
    mrow[ji] = 0.f;
  }
}

// ---------------------------------------------------------------------------
extern "C" void kernel_launch(void* const* d_in, const int* in_sizes, int n_in,
                              void* d_out, int out_size, void* d_ws, size_t ws_size,
                              hipStream_t stream) {
  (void)in_sizes; (void)n_in; (void)out_size; (void)ws_size;
  const float* x  = (const float*)d_in[0];
  // d_in[1] is k (==10, hardcoded)
  const float* w1 = (const float*)d_in[2];
  const float* g1 = (const float*)d_in[3];
  const float* b1 = (const float*)d_in[4];
  const float* m1 = (const float*)d_in[5];
  const float* v1 = (const float*)d_in[6];
  const float* w2 = (const float*)d_in[7];
  const float* g2 = (const float*)d_in[8];
  const float* b2 = (const float*)d_in[9];
  const float* m2 = (const float*)d_in[10];
  const float* v2 = (const float*)d_in[11];

  char* ws = (char*)d_ws;
  size_t off = 0;
  auto wsalloc = [&](size_t bytes) -> void* {
    void* p = ws + off;
    off = (off + bytes + 255) & ~(size_t)255;
    return p;
  };
  _Float16* Xh = (_Float16*)wsalloc((size_t)NB * NN * CC * 2);   //  4 MB
  float*    sq = (float*)   wsalloc((size_t)NB * NN * 4);        //  0.13 MB
  int*      kn = (int*)     wsalloc((size_t)NB * NN * KNN * 4);  //  1.3 MB
  float*    P  = (float*)   wsalloc((size_t)NB * NN * CC * 4);   //  8 MB
  float*    Q  = (float*)   wsalloc((size_t)NB * NN * CC * 4);   //  8 MB
  float*    ev = (float*)   wsalloc((size_t)NB * NN * KNN * 4);  //  1.3 MB

  float* xout = (float*)d_out;                       // (B,N,C)
  float* eij  = xout + (size_t)NB * NN * CC;         // (B,N,N)
  float* mask = eij  + (size_t)NB * NN * NN;         // (B,N,N)

  pack_kernel   <<<(NB * NN) / 256, 256, 0, stream>>>(x, Xh, sq);
  knn_kernel    <<<(NB * NN / 16) / 4, 128, 0, stream>>>(Xh, sq, kn);
  pq_kernel     <<<(2 * NB * (NN / 16) * (CC / 16)) / 4, 128, 0, stream>>>(Xh, w1, P, Q);
  point_kernel  <<<(NB * NN) / 256, 256, 0, stream>>>(P, Q, kn, g1, b1, m1, v1,
                                                      w2, g2, b2, m2, v2, xout, ev);
  softmax_kernel<<<(NB * NN) / 8, 256, 0, stream>>>(ev, kn, eij, mask);
}
